// RNNEncoder_38517266710835
// MI455X (gfx1250) — matrix operand order repaired and checked by reading, not compiled, not run on hardware
//
#include <hip/hip_runtime.h>

typedef __bf16 bf16;
typedef bf16  v16bf __attribute__((ext_vector_type(16)));
typedef float v8f   __attribute__((ext_vector_type(8)));

#define T_STEPS 64
#define BATCH   512
#define IN_DIM  1024
#define DNN     256
#define MEM     64
#define LAYERS  2
#define HEADS   8
#define DH      32
#define TRAN    2
#define ROWS    16
#define ATTN_BIAS 5.0f

struct alignas(16) U4 { unsigned a, b, c, d; };
struct U8 { U4 lo, hi; };

__device__ __forceinline__ float asf(unsigned u) { return __builtin_bit_cast(float, u); }
__device__ __forceinline__ float bflo(unsigned u) { return __builtin_bit_cast(float, u << 16); }
__device__ __forceinline__ float bfhi(unsigned u) { return __builtin_bit_cast(float, u & 0xffff0000u); }

// native RNE f32->bf16 (hardware packed convert)
__device__ __forceinline__ bf16 f2bf(float f) { return (bf16)f; }
__device__ __forceinline__ unsigned pack2(float a, float b) {
  unsigned short ha = __builtin_bit_cast(unsigned short, (bf16)a);
  unsigned short hb = __builtin_bit_cast(unsigned short, (bf16)b);
  return (unsigned)ha | ((unsigned)hb << 16);
}
// 8 f32 (two U4) * scale -> 8 packed bf16 (one U4)
__device__ __forceinline__ U4 pack8(U4 f0, U4 f1, float s) {
  U4 o;
  o.a = pack2(asf(f0.a) * s, asf(f0.b) * s);
  o.b = pack2(asf(f0.c) * s, asf(f0.d) * s);
  o.c = pack2(asf(f1.a) * s, asf(f1.b) * s);
  o.d = pack2(asf(f1.c) * s, asf(f1.d) * s);
  return o;
}
__device__ __forceinline__ float dot8(U4 u, const float* q) {
  return bflo(u.a) * q[0] + bfhi(u.a) * q[1] + bflo(u.b) * q[2] + bfhi(u.b) * q[3] +
         bflo(u.c) * q[4] + bfhi(u.c) * q[5] + bflo(u.d) * q[6] + bfhi(u.d) * q[7];
}
__device__ __forceinline__ void fma8(U4 u, float p, float* o) {
  o[0] += p * bflo(u.a); o[1] += p * bfhi(u.a);
  o[2] += p * bflo(u.b); o[3] += p * bfhi(u.b);
  o[4] += p * bflo(u.c); o[5] += p * bfhi(u.c);
  o[6] += p * bflo(u.d); o[7] += p * bfhi(u.d);
}
__device__ __forceinline__ v16bf ld2(const bf16* p0, const bf16* p1) {
  U8 u; u.lo = *(const U4*)p0; u.hi = *(const U4*)p1;
  return __builtin_bit_cast(v16bf, u);
}
// A fragment from an f32 LDS tile (row stride 32), converting to bf16 in regs
__device__ __forceinline__ v16bf ldfrag_f32(const float* p) {
  U4 a = *(const U4*)p,        b = *(const U4*)(p + 4);
  U4 c = *(const U4*)(p + 16), d = *(const U4*)(p + 20);
  U8 u;
  u.lo = pack8(a, b, 1.f);
  u.hi = pack8(c, d, 1.f);
  return __builtin_bit_cast(v16bf, u);
}
__device__ __forceinline__ v8f wmma_bf16(v16bf a, v16bf b, v8f c) {
  return __builtin_amdgcn_wmma_f32_16x16x32_bf16(false, a, false, b, (short)0, c,
                                                 false, false);
}
__device__ __forceinline__ float sigmoidf(float x) { return 1.0f / (1.0f + __expf(-x)); }

// CDNA5 async global->LDS copy (ASYNCcnt-tracked), 16B per lane
__device__ __forceinline__ void async_cp16(void* lds_dst, const void* gsrc) {
  unsigned loff = (unsigned)(size_t)lds_dst;
  unsigned long long ga = (unsigned long long)(size_t)gsrc;
  asm volatile("global_load_async_to_lds_b128 %0, %1, off" :: "v"(loff), "v"(ga)
               : "memory");
}
__device__ __forceinline__ void async_wait0() {
  asm volatile("s_wait_asynccnt 0x0" ::: "memory");
}

// ---------------------------------------------------------------------------
// Transpose + f32->bf16 convert:  dst[c*rows + r] = bf16(src[r*cols + c])
// ---------------------------------------------------------------------------
__global__ void k_convert_tr(const float* __restrict__ src, bf16* __restrict__ dst,
                             int rows, int cols) {
  int n = rows * cols;
  for (int i = blockIdx.x * blockDim.x + threadIdx.x; i < n;
       i += gridDim.x * blockDim.x) {
    int r = i / cols, c = i - r * cols;
    dst[(size_t)c * rows + r] = f2bf(src[i]);
  }
}

// ---------------------------------------------------------------------------
// Input FC: xe = relu(x @ W_in + b_in). A tile (64x32 f32) staged with async
// global->LDS copies, double-buffered; bf16 conversion folded into frag load.
// ---------------------------------------------------------------------------
__device__ __forceinline__ void stage_tile(const float* __restrict__ x, int rows0,
                                           int kk, float* dst, int tid) {
  int i0 = tid * 8;  // 2048 f32 per tile / 256 thr
  const float* src = &x[(size_t)(rows0 + (i0 >> 5)) * IN_DIM + kk * 32 + (i0 & 31)];
  async_cp16(dst + i0, src);
  async_cp16(dst + i0 + 4, src + 4);
}

__global__ void __launch_bounds__(256) k_in_fc(const float* __restrict__ x,
                                               const bf16* __restrict__ Wt,
                                               const float* __restrict__ bias,
                                               bf16* __restrict__ xe) {
  __shared__ float aS[2][64 * 32];  // double-buffered async-filled A tile
  const int tid = threadIdx.x;
  const int lane = tid & 31, w = tid >> 5;
  const int arow = lane & 15, akb = (lane >> 4) << 3;
  const int bcol = lane & 15, bkh = (lane >> 4) << 3;
  const int rt = w & 3, cb = (w >> 2) * 128;
  const int rows0 = blockIdx.x * 64;
  v8f acc[8] = {};

  stage_tile(x, rows0, 0, aS[0], tid);
  async_wait0();
  __syncthreads();
  int buf = 0;
  for (int kk = 0; kk < IN_DIM / 32; ++kk) {
    if (kk + 1 < IN_DIM / 32) stage_tile(x, rows0, kk + 1, aS[buf ^ 1], tid);
    v16bf af = ldfrag_f32(&aS[buf][(rt * 16 + arow) * 32 + akb]);
#pragma unroll
    for (int jt = 0; jt < 8; ++jt) {
      const bf16* bp = Wt + (size_t)(cb + jt * 16 + bcol) * IN_DIM + kk * 32 + bkh;
      acc[jt] = wmma_bf16(af, ld2(bp, bp + 16), acc[jt]);
    }
    async_wait0();
    __syncthreads();
    buf ^= 1;
  }
  const int mbase = (lane >> 4) << 3, ncol = lane & 15;
#pragma unroll
  for (int jt = 0; jt < 8; ++jt) {
    int d = cb + jt * 16 + ncol;
#pragma unroll
    for (int j = 0; j < 8; ++j) {
      int m = mbase + j;
      float v = acc[jt][j] + bias[d];
      v = v > 0.f ? v : 0.f;
      xe[(size_t)(rows0 + rt * 16 + m) * DNN + d] = f2bf(v);
    }
  }
}

// ---------------------------------------------------------------------------
// Output FC: out = relu(hout @ W_out + b_out); A frags direct from global bf16.
// ---------------------------------------------------------------------------
__global__ void __launch_bounds__(256) k_out_fc(const bf16* __restrict__ hout,
                                                const bf16* __restrict__ Wt,
                                                const float* __restrict__ bias,
                                                float* __restrict__ out) {
  const int tid = threadIdx.x;
  const int lane = tid & 31, w = tid >> 5;
  const int arow = lane & 15, akb = (lane >> 4) << 3;
  const int bcol = lane & 15, bkh = (lane >> 4) << 3;
  const int rt = w & 3, cb = (w >> 2) * 128;
  const int rows0 = blockIdx.x * 64;
  v8f acc[8] = {};
  for (int kk = 0; kk < DNN / 32; ++kk) {
    const bf16* ap = hout + (size_t)(rows0 + rt * 16 + arow) * DNN + kk * 32 + akb;
    v16bf af = ld2(ap, ap + 16);
#pragma unroll
    for (int jt = 0; jt < 8; ++jt) {
      const bf16* bp = Wt + (size_t)(cb + jt * 16 + bcol) * DNN + kk * 32 + bkh;
      acc[jt] = wmma_bf16(af, ld2(bp, bp + 16), acc[jt]);
    }
  }
  const int mbase = (lane >> 4) << 3, ncol = lane & 15;
#pragma unroll
  for (int jt = 0; jt < 8; ++jt) {
    int d = cb + jt * 16 + ncol;
#pragma unroll
    for (int j = 0; j < 8; ++j) {
      int m = mbase + j;
      float v = acc[jt][j] + bias[d];
      out[(size_t)(rows0 + rt * 16 + m) * DNN + d] = v > 0.f ? v : 0.f;
    }
  }
}

// ---------------------------------------------------------------------------
// Persistent recurrent kernel: block owns 16 batch rows for the whole scan.
// Wave w is output-stationary on d-slice [32w,32w+32) of i,f,o,g,a,k,v,q;
// attention head w == wave w slice. Ring buffer memory (bf16) in global ws.
// 4 barriers per cell; attention packs bf16 straight into the GEMM2 A tile.
// ---------------------------------------------------------------------------
__global__ void __launch_bounds__(256) k_recurrent(
    const bf16* __restrict__ xe, const unsigned char* __restrict__ done,
    const float* __restrict__ h0, const float* __restrict__ c0,
    const bf16* __restrict__ Wt_main, const bf16* __restrict__ Wt_attn,
    const bf16* __restrict__ Wt_ao, const float* __restrict__ b_main,
    bf16* __restrict__ memk, bf16* __restrict__ memv, bf16* __restrict__ hout) {
  extern __shared__ char smem[];
  float* hS = (float*)smem;                         // [L][16][256]
  float* cS = hS + LAYERS * ROWS * DNN;             // [L][16][256]
  bf16* combS = (bf16*)(cS + LAYERS * ROWS * DNN);  // [16][512]
  float* qS = (float*)(combS + ROWS * 2 * DNN);     // [16][256]
  int* cntS = (int*)(qS + ROWS * DNN);              // [16]
  float* ndS = (float*)(cntS + ROWS);               // [16]

  const int tid = threadIdx.x;
  const int lane = tid & 31, w = tid >> 5;
  const int arow = lane & 15, akb = (lane >> 4) << 3;
  const int bcol = lane & 15, bkh = (lane >> 4) << 3;
  const int mbase = (lane >> 4) << 3, ncol = lane & 15;
  const int rowbase = blockIdx.x * ROWS;

  // vectorized state init
  for (int i0 = tid * 4; i0 < LAYERS * ROWS * DNN; i0 += 256 * 4) {
    int l = i0 / (ROWS * DNN);
    int rem = i0 - l * ROWS * DNN;
    int r = rem >> 8, d = rem & 255;
    size_t g = ((size_t)l * BATCH + rowbase + r) * DNN + d;
    *(U4*)&hS[i0] = *(const U4*)&h0[g];
    *(U4*)&cS[i0] = *(const U4*)&c0[g];
  }
  if (tid < ROWS) cntS[tid] = 0;
  __syncthreads();

  for (int t = 0; t < T_STEPS; ++t) {
    for (int tt = 0; tt < TRAN; ++tt) {
      const int step = t * TRAN + tt;
      const int widx = step & (MEM - 1);
      // per-row bookkeeping: fold previous step's append count into this setup
      if (tid < ROWS) {
        int c = cntS[tid] + (step > 0 ? 1 : 0);
        float nd = 1.f;
        if (tt == 0 && done[t * BATCH + rowbase + tid]) { nd = 0.f; c = 0; }
        cntS[tid] = c;
        ndS[tid] = nd;
      }
      __syncthreads();

      for (int l = 0; l < LAYERS; ++l) {
        // ---- build comb = [xin, h*nd] in bf16 (16B-vectorized) ----
        if (l == 0) {
          // async copy xe rows straight into LDS (bf16 already)
          for (int ci = tid; ci < ROWS * DNN / 8; ci += 256) {
            int r = ci >> 5, e0 = (ci & 31) * 8;
            async_cp16(&combS[r * 512 + e0],
                       &xe[((size_t)t * BATCH + rowbase + r) * DNN + e0]);
          }
        } else {
          for (int ci = tid; ci < ROWS * DNN / 8; ci += 256) {
            int r = ci >> 5, e0 = (ci & 31) * 8;
            const float* hp = &hS[r * DNN + e0];  // layer-0 fresh h
            *(U4*)&combS[r * 512 + e0] =
                pack8(*(const U4*)hp, *(const U4*)(hp + 4), 1.f);
          }
        }
        for (int ci = tid; ci < ROWS * DNN / 8; ci += 256) {
          int r = ci >> 5, e0 = (ci & 31) * 8;
          const float* hp = &hS[(l * ROWS + r) * DNN + e0];
          *(U4*)&combS[r * 512 + DNN + e0] =
              pack8(*(const U4*)hp, *(const U4*)(hp + 4), ndS[r]);
        }
        if (l == 0) async_wait0();
        __syncthreads();

        // ---- GEMM1: comb(16x512) @ [W_main|W_attn](512x2048) ----
        const bf16* Wm = Wt_main + (size_t)l * 1280 * 512;
        const bf16* Wa = Wt_attn + (size_t)l * 768 * 512;
        v8f acc[16] = {};
        for (int kk = 0; kk < 16; ++kk) {
          const int k0 = kk * 32;
          const bf16* ap = combS + arow * 512 + k0 + akb;
          v16bf af = ld2(ap, ap + 16);
          __builtin_prefetch(Wm + (size_t)(w * 32 + bcol) * 512 + k0 + 32, 0, 0);
#pragma unroll
          for (int g = 0; g < 8; ++g) {
#pragma unroll
            for (int ts = 0; ts < 2; ++ts) {
              int cg = g * DNN + w * 32 + ts * 16;
              const bf16* bp = (g < 5)
                  ? Wm + (size_t)(cg + bcol) * 512 + k0 + bkh
                  : Wa + (size_t)(cg - 5 * DNN + bcol) * 512 + k0 + bkh;
              acc[g * 2 + ts] = wmma_bf16(af, ld2(bp, bp + 16), acc[g * 2 + ts]);
            }
          }
        }

        // ---- epilogue: gates, k/v to ring, q to LDS ----
        float cnA[16], avA[16], ovA[16];
        const float* bm = b_main + (size_t)l * 1280;
#pragma unroll
        for (int ts = 0; ts < 2; ++ts) {
#pragma unroll
          for (int j = 0; j < 8; ++j) {
            int m = mbase + j;
            int d = w * 32 + ts * 16 + ncol;
            float ndv = ndS[m];
            float iv = sigmoidf(acc[0 * 2 + ts][j] + bm[0 * DNN + d]);
            float fv = sigmoidf(acc[1 * 2 + ts][j] + bm[1 * DNN + d]);
            float ov = sigmoidf(acc[2 * 2 + ts][j] + bm[2 * DNN + d]);
            float gv = tanhf(acc[3 * 2 + ts][j] + bm[3 * DNN + d]);
            float av = sigmoidf(acc[4 * 2 + ts][j] + bm[4 * DNN + d]);
            float kv = acc[5 * 2 + ts][j];
            float vv = acc[6 * 2 + ts][j];
            float qv = acc[7 * 2 + ts][j];
            cnA[ts * 8 + j] = fv * (cS[(l * ROWS + m) * DNN + d] * ndv) + iv * gv;
            avA[ts * 8 + j] = av;
            ovA[ts * 8 + j] = ov;
            qS[m * DNN + d] = qv;
            size_t ridx = (((size_t)l * BATCH + rowbase + m) * MEM + widx) * DNN + d;
            memk[ridx] = f2bf(kv);
            memv[ridx] = f2bf(vv);
          }
        }
        __threadfence_block();  // ring only read within this workgroup (WGP scope)
        __syncthreads();

        // ---- attention: one thread per (row, head); online softmax; packs its
        //      contiguous 32-elem ao slice as bf16 straight into the GEMM2 A tile
        if (tid < ROWS * HEADS) {
          int r = tid >> 3, hd = tid & 7;
          int valid = cntS[r] + 1;
          if (valid > MEM) valid = MEM;
          const size_t base = ((size_t)l * BATCH + rowbase + r) * MEM;
          const bf16* kb = memk + base * DNN + hd * DH;
          const bf16* vb = memv + base * DNN + hd * DH;
          float q[DH], ov[DH];
#pragma unroll
          for (int jj = 0; jj < DH; jj += 4) {
            U4 qw = *(const U4*)&qS[r * DNN + hd * DH + jj];
            q[jj] = asf(qw.a); q[jj + 1] = asf(qw.b);
            q[jj + 2] = asf(qw.c); q[jj + 3] = asf(qw.d);
          }
#pragma unroll
          for (int jj = 0; jj < DH; ++jj) ov[jj] = 0.f;
          float mx = -3.4e38f, sm = 0.f;
          for (int age = 0; age < valid; ++age) {
            int slot = widx - age;
            slot += (slot >> 31) & MEM;
            const U4* kp = (const U4*)(kb + (size_t)slot * DNN);
            U4 ka = kp[0], kb4 = kp[1], kc = kp[2], kd = kp[3];
            float s = dot8(ka, q) + dot8(kb4, q + 8) + dot8(kc, q + 16) +
                      dot8(kd, q + 24);
            s *= 0.1767766952966369f;  // 1/sqrt(32)
            if (age == 0) s += ATTN_BIAS;
            float nmx = fmaxf(mx, s);
            float corr = __expf(mx - nmx);
            float p = __expf(s - nmx);
            sm = sm * corr + p;
            const U4* vp = (const U4*)(vb + (size_t)slot * DNN);
            U4 va = vp[0], vb4 = vp[1], vc = vp[2], vd = vp[3];
#pragma unroll
            for (int jj = 0; jj < DH; ++jj) ov[jj] *= corr;
            fma8(va, p, ov); fma8(vb4, p, ov + 8);
            fma8(vc, p, ov + 16); fma8(vd, p, ov + 24);
            mx = nmx;
          }
          float inv = 1.f / sm;
          bf16* outp = &combS[r * DNN + hd * DH];
#pragma unroll
          for (int jj = 0; jj < DH; jj += 8) {
            U4 o;
            o.a = pack2(ov[jj + 0] * inv, ov[jj + 1] * inv);
            o.b = pack2(ov[jj + 2] * inv, ov[jj + 3] * inv);
            o.c = pack2(ov[jj + 4] * inv, ov[jj + 5] * inv);
            o.d = pack2(ov[jj + 6] * inv, ov[jj + 7] * inv);
            *(U4*)(outp + jj) = o;
          }
        }
        __syncthreads();

        // ---- GEMM2: ao(16x256) @ W_ao(256x256), wave d-slice ----
        const bf16* Wo = Wt_ao + (size_t)l * DNN * DNN;
        v8f acc2[2] = {};
        for (int kk = 0; kk < 8; ++kk) {
          const int k0 = kk * 32;
          const bf16* ap = combS + arow * DNN + k0 + akb;
          v16bf af = ld2(ap, ap + 16);
#pragma unroll
          for (int ts = 0; ts < 2; ++ts) {
            const bf16* bp = Wo + (size_t)(w * 32 + ts * 16 + bcol) * DNN + k0 + bkh;
            acc2[ts] = wmma_bf16(af, ld2(bp, bp + 16), acc2[ts]);
          }
        }

        // ---- finalize state (no barrier needed between GEMM2 and here) ----
#pragma unroll
        for (int ts = 0; ts < 2; ++ts) {
#pragma unroll
          for (int j = 0; j < 8; ++j) {
            int m = mbase + j;
            int d = w * 32 + ts * 16 + ncol;
            float cfin = cnA[ts * 8 + j] + avA[ts * 8 + j] * tanhf(acc2[ts][j]);
            float hfin = ovA[ts * 8 + j] * tanhf(cfin);
            cS[(l * ROWS + m) * DNN + d] = cfin;
            hS[(l * ROWS + m) * DNN + d] = hfin;
            if (l == LAYERS - 1 && tt == TRAN - 1)
              hout[((size_t)t * BATCH + rowbase + m) * DNN + d] = f2bf(hfin);
          }
        }
        __syncthreads();
      }  // l
    }  // tt
  }  // t
}

// ---------------------------------------------------------------------------
extern "C" void kernel_launch(void* const* d_in, const int* in_sizes, int n_in,
                              void* d_out, int out_size, void* d_ws, size_t ws_size,
                              hipStream_t stream) {
  (void)in_sizes; (void)n_in; (void)out_size; (void)ws_size;
  const float* x = (const float*)d_in[0];
  const unsigned char* done = (const unsigned char*)d_in[1];
  const float* h0 = (const float*)d_in[2];
  const float* c0 = (const float*)d_in[3];
  const float* W_in = (const float*)d_in[7];
  const float* b_in = (const float*)d_in[8];
  const float* W_main = (const float*)d_in[9];
  const float* b_main = (const float*)d_in[10];
  const float* W_attn = (const float*)d_in[11];
  const float* W_ao = (const float*)d_in[12];
  const float* W_out = (const float*)d_in[13];
  const float* b_out = (const float*)d_in[14];

  char* ws = (char*)d_ws;
  size_t off = 0;
  bf16* Wt_in = (bf16*)(ws + off);   off += (size_t)IN_DIM * DNN * 2;
  bf16* Wt_main = (bf16*)(ws + off); off += (size_t)LAYERS * 1280 * 512 * 2;
  bf16* Wt_attn = (bf16*)(ws + off); off += (size_t)LAYERS * 768 * 512 * 2;
  bf16* Wt_ao = (bf16*)(ws + off);   off += (size_t)LAYERS * DNN * DNN * 2;
  bf16* Wt_out = (bf16*)(ws + off);  off += (size_t)DNN * DNN * 2;
  bf16* xe = (bf16*)(ws + off);      off += (size_t)T_STEPS * BATCH * DNN * 2;
  bf16* hout = (bf16*)(ws + off);    off += (size_t)T_STEPS * BATCH * DNN * 2;
  bf16* memk = (bf16*)(ws + off);    off += (size_t)LAYERS * BATCH * MEM * DNN * 2;
  bf16* memv = (bf16*)(ws + off);    off += (size_t)LAYERS * BATCH * MEM * DNN * 2;

  k_convert_tr<<<256, 256, 0, stream>>>(W_in, Wt_in, IN_DIM, DNN);
  k_convert_tr<<<256, 256, 0, stream>>>(W_main, Wt_main, 512, 1280);
  k_convert_tr<<<256, 256, 0, stream>>>(W_main + 512 * 1280, Wt_main + 1280 * 512, 512, 1280);
  k_convert_tr<<<256, 256, 0, stream>>>(W_attn, Wt_attn, 512, 768);
  k_convert_tr<<<256, 256, 0, stream>>>(W_attn + 512 * 768, Wt_attn + 768 * 512, 512, 768);
  k_convert_tr<<<256, 256, 0, stream>>>(W_ao, Wt_ao, DNN, DNN);
  k_convert_tr<<<256, 256, 0, stream>>>(W_ao + DNN * DNN, Wt_ao + DNN * DNN, DNN, DNN);
  k_convert_tr<<<256, 256, 0, stream>>>(W_out, Wt_out, DNN, DNN);

  k_in_fc<<<T_STEPS * BATCH / 64, 256, 0, stream>>>(x, Wt_in, b_in, xe);

  size_t smem = (size_t)(LAYERS * ROWS * DNN * 4) * 2  // hS + cS
              + (size_t)ROWS * 2 * DNN * 2             // combS
              + (size_t)ROWS * DNN * 4                 // qS
              + (size_t)ROWS * (4 + 4);                // cntS + ndS
  k_recurrent<<<BATCH / ROWS, 256, smem, stream>>>(xe, done, h0, c0, Wt_main, Wt_attn,
                                                   Wt_ao, b_main, memk, memv, hout);

  k_out_fc<<<T_STEPS * BATCH / 64, 256, 0, stream>>>(hout, Wt_out, b_out, (float*)d_out);
}
